// LinearAttention_8624294330868
// MI455X (gfx1250) — compile-verified, matrix operand order checked by reading, not run on hardware
//
#include <hip/hip_runtime.h>
#include <hip/hip_bf16.h>

// ---------------------------------------------------------------------------
// LinearAttention for MI455X (gfx1250): bf16 WMMA, all fragment operands as
// contiguous 128-bit loads (two 8-element runs per lane, per ISA layout).
//   b=8, n=8192, C=256, heads=8, dh=32
// ---------------------------------------------------------------------------

typedef __attribute__((ext_vector_type(16))) __bf16 v16bf;
typedef __attribute__((ext_vector_type(8)))  __bf16 v8bf;
typedef __attribute__((ext_vector_type(8)))  float  v8f;

union AFrag { v16bf v; v8bf h[2]; __bf16 e[16]; };
union CFrag { v8f   v; float  e[8]; };
union BF8   { v8bf  v; __bf16 e[8]; };

#define BATCH 8
#define SEQ   8192
#define CDIM  256
#define NHEAD 8
#define DH    32
#define NROWS (BATCH * SEQ)          // 65536
#define ATT_SCALE 0.0625f            // 256^-0.5
#define LN_EPS 1e-5f

// Workspace layout (bytes)
#define OFF_Q   ((size_t)0)                             // qT  [8][256][8192] bf16
#define OFF_K   (OFF_Q  + (size_t)NROWS * CDIM * 2)     // kT  [8][256][8192] f32
#define OFF_V   (OFF_K  + (size_t)NROWS * CDIM * 4)     // v   [65536][256]   bf16
#define OFF_M   (OFF_V  + (size_t)NROWS * CDIM * 2)     // max [8*256] f32
#define OFF_Z   (OFF_M  + (size_t)BATCH * CDIM * 4)     // 1/Z [8*256] f32
#define OFF_A   (OFF_Z  + (size_t)BATCH * CDIM * 4)     // A   [64][32][32] f32
#define OFF_WQT (OFF_A  + (size_t)64 * 1024 * 4)        // WqkvT [768][256] bf16
#define OFF_WOT (OFF_WQT + (size_t)768 * 256 * 2)       // WoutT [256][256] bf16

// Lane's 16 fragment elements = runs K in {hi8..hi8+7} and {16+hi8..23+hi8}
__device__ __forceinline__ int frag_k_idx(int j, int half) {
    return j + ((j < 8) ? 0 : 8) + half * 8;
}

__device__ __forceinline__ void cvt8(const float* __restrict__ p, __bf16* d) {
    const float4 u0 = *(const float4*)p;
    const float4 u1 = *(const float4*)(p + 4);
    d[0] = (__bf16)u0.x; d[1] = (__bf16)u0.y; d[2] = (__bf16)u0.z; d[3] = (__bf16)u0.w;
    d[4] = (__bf16)u1.x; d[5] = (__bf16)u1.y; d[6] = (__bf16)u1.z; d[7] = (__bf16)u1.w;
}

__device__ __forceinline__ void exp8(const float* __restrict__ p, float m, float z,
                                     __bf16* d) {
    const float4 u0 = *(const float4*)p;
    const float4 u1 = *(const float4*)(p + 4);
    d[0] = (__bf16)(__expf(u0.x - m) * z); d[1] = (__bf16)(__expf(u0.y - m) * z);
    d[2] = (__bf16)(__expf(u0.z - m) * z); d[3] = (__bf16)(__expf(u0.w - m) * z);
    d[4] = (__bf16)(__expf(u1.x - m) * z); d[5] = (__bf16)(__expf(u1.y - m) * z);
    d[6] = (__bf16)(__expf(u1.z - m) * z); d[7] = (__bf16)(__expf(u1.w - m) * z);
}

// ---------------------------------------------------------------------------
// K0: one-time weight transposes to bf16 (L2-resident B operands).
// ---------------------------------------------------------------------------
__global__ __launch_bounds__(256) void k_transpose(const float* __restrict__ Wqkv,
                                                   const float* __restrict__ Wout,
                                                   __bf16* __restrict__ WqkvT,
                                                   __bf16* __restrict__ WoutT) {
    int tid = blockIdx.x * 256 + threadIdx.x;
    if (tid < 256 * 768) {
        int r = tid / 768, c = tid % 768;
        WqkvT[c * 256 + r] = (__bf16)Wqkv[tid];
    }
    if (tid < 256 * 256) {
        int r = tid >> 8, c = tid & 255;
        WoutT[c * 256 + r] = (__bf16)Wout[tid];
    }
}

// ---------------------------------------------------------------------------
// K1: qkv = x[65536,256] @ Wqkv.  No LDS, no barriers: A-frags are two b128
// loads from x rows (cvt to bf16), B-frags two b128 loads from WqkvT rows.
// q,k written transposed (vectorized stores); v row-major for K4.
// ---------------------------------------------------------------------------
__global__ __launch_bounds__(256) void k_qkv_gemm(const float* __restrict__ x,
                                                  const __bf16* __restrict__ WqkvT,
                                                  __bf16* __restrict__ qT,
                                                  float*  __restrict__ kT,
                                                  __bf16* __restrict__ vb) {
    const int tid = threadIdx.x, lane = tid & 31, wv = tid >> 5;
    const int l15 = lane & 15, half = lane >> 4, hi8 = half * 8;
    const int mbase = blockIdx.x * 128;
    const int nbase = blockIdx.y * 64;

    const float* xrow = x + (size_t)(mbase + wv * 16 + l15) * CDIM;

    CFrag acc[4];
    for (int i = 0; i < 4; ++i)
        for (int r = 0; r < 8; ++r) acc[i].e[r] = 0.0f;

    for (int k0 = 0; k0 < CDIM; k0 += 32) {
        if (k0 + 32 < CDIM) __builtin_prefetch(xrow + k0 + 32, 0, 0);
        AFrag a;
        cvt8(xrow + k0 + hi8,      &a.e[0]);
        cvt8(xrow + k0 + 16 + hi8, &a.e[8]);
        for (int nt = 0; nt < 4; ++nt) {
            const __bf16* wrow = WqkvT + (size_t)(nbase + nt * 16 + l15) * CDIM + k0;
            AFrag bf;
            bf.h[0] = *(const v8bf*)(wrow + hi8);
            bf.h[1] = *(const v8bf*)(wrow + 16 + hi8);
            acc[nt].v = __builtin_amdgcn_wmma_f32_16x16x32_bf16(
                false, a.v, false, bf.v, (short)0, acc[nt].v, false, false);
        }
    }

    const int seg     = nbase >> 8;   // 0=q, 1=k, 2=v (uniform per WG)
    const int colbase = nbase & 255;
    const int b    = mbase >> 13;
    const int nloc = (mbase & (SEQ - 1)) + wv * 16 + hi8;   // n for r=0 (mult of 8)

    if (seg == 0) {          // q -> qT[b][c][n], one 16B store per tile
        for (int nt = 0; nt < 4; ++nt) {
            int c = colbase + nt * 16 + l15;
            BF8 pk;
            for (int r = 0; r < 8; ++r) pk.e[r] = (__bf16)acc[nt].e[r];
            *(v8bf*)(qT + (size_t)(b * CDIM + c) * SEQ + nloc) = pk.v;
        }
    } else if (seg == 1) {   // k -> kT[b][c][n] f32, two 16B stores per tile
        for (int nt = 0; nt < 4; ++nt) {
            int c = colbase + nt * 16 + l15;
            float* dst = kT + (size_t)(b * CDIM + c) * SEQ + nloc;
            float4 s0 = {acc[nt].e[0], acc[nt].e[1], acc[nt].e[2], acc[nt].e[3]};
            float4 s1 = {acc[nt].e[4], acc[nt].e[5], acc[nt].e[6], acc[nt].e[7]};
            *(float4*)dst = s0;
            *(float4*)(dst + 4) = s1;
        }
    } else {                 // v -> row-major [row][256] bf16
        for (int nt = 0; nt < 4; ++nt) {
            int c = colbase + nt * 16 + l15;
            for (int r = 0; r < 8; ++r) {
                int grow = b * SEQ + nloc + r;
                vb[(size_t)grow * CDIM + c] = (__bf16)acc[nt].e[r];
            }
        }
    }
}

// ---------------------------------------------------------------------------
// K2: softmax stats per (b, channel) row of kT: online max/sum-exp with
// wave shfl reduction + tiny LDS combine.  2048 blocks, coalesced reads.
// ---------------------------------------------------------------------------
__global__ __launch_bounds__(256) void k_stats(const float* __restrict__ kT,
                                               float* __restrict__ ms,
                                               float* __restrict__ rz) {
    __shared__ float wm[8], wsum[8];
    const int row = blockIdx.x;                 // b*256 + c
    const float* p = kT + (size_t)row * SEQ;
    const int tid = threadIdx.x;
    float m = -3.4e38f, s = 0.0f;
    for (int i = tid; i < SEQ; i += 256) {
        float v = p[i];
        if (v > m) { s = s * __expf(m - v) + 1.0f; m = v; }
        else       { s += __expf(v - m); }
    }
    for (int off = 16; off > 0; off >>= 1) {
        float mo = __shfl_xor(m, off, 32);
        float so = __shfl_xor(s, off, 32);
        float mn = fmaxf(m, mo);
        s = s * __expf(m - mn) + so * __expf(mo - mn);
        m = mn;
    }
    if ((tid & 31) == 0) { wm[tid >> 5] = m; wsum[tid >> 5] = s; }
    __syncthreads();
    if (tid == 0) {
        float M = wm[0], S = wsum[0];
        for (int j = 1; j < 8; ++j) {
            float mn = fmaxf(M, wm[j]);
            S = S * __expf(M - mn) + wsum[j] * __expf(wm[j] - mn);
            M = mn;
        }
        ms[row] = M;
        rz[row] = 1.0f / S;
    }
}

// ---------------------------------------------------------------------------
// K3: A[b,h] = scale * softmax(k)^T(32x8192) @ q(8192x32).
// One WG per (b,h); 8 waves own disjoint 1024-wide K chunks and compute all
// four 16x16 tiles; exp applied in registers from kT rows (b128 loads);
// q B-frags are b128 loads from qT rows.  ds_add_f32 reduction.
// ---------------------------------------------------------------------------
__global__ __launch_bounds__(256) void k_amat(const float* __restrict__ kT,
                                              const __bf16* __restrict__ qT,
                                              const float* __restrict__ ms,
                                              const float* __restrict__ rz,
                                              float* __restrict__ Am) {
    __shared__ float red[4][256];
    const int bh = blockIdx.x, b = bh >> 3, h = bh & 7;
    const int tid = threadIdx.x, lane = tid & 31, wv = tid >> 5;
    const int l15 = lane & 15, half = lane >> 4, hi8 = half * 8;

    for (int i = tid; i < 1024; i += 256) red[i >> 8][i & 255] = 0.0f;
    __syncthreads();

    const int c0 = h * DH + l15, c1 = c0 + 16;
    const float m0 = ms[b * CDIM + c0], z0 = rz[b * CDIM + c0];
    const float m1 = ms[b * CDIM + c1], z1 = rz[b * CDIM + c1];
    const float*  k0p = kT + (size_t)(b * CDIM + c0) * SEQ;
    const float*  k1p = kT + (size_t)(b * CDIM + c1) * SEQ;
    const __bf16* q0p = qT + (size_t)(b * CDIM + c0) * SEQ;
    const __bf16* q1p = qT + (size_t)(b * CDIM + c1) * SEQ;

    CFrag acc[4];
    for (int i = 0; i < 4; ++i)
        for (int r = 0; r < 8; ++r) acc[i].e[r] = 0.0f;

    const int i0 = wv * (SEQ / 8);
    for (int s = 0; s < SEQ / 8; s += 32) {
        const int base = i0 + s;
        __builtin_prefetch(k0p + base + 32, 0, 0);
        AFrag a0, a1, b0, b1;
        exp8(k0p + base + hi8,      m0, z0, &a0.e[0]);
        exp8(k0p + base + 16 + hi8, m0, z0, &a0.e[8]);
        exp8(k1p + base + hi8,      m1, z1, &a1.e[0]);
        exp8(k1p + base + 16 + hi8, m1, z1, &a1.e[8]);
        b0.h[0] = *(const v8bf*)(q0p + base + hi8);
        b0.h[1] = *(const v8bf*)(q0p + base + 16 + hi8);
        b1.h[0] = *(const v8bf*)(q1p + base + hi8);
        b1.h[1] = *(const v8bf*)(q1p + base + 16 + hi8);
        acc[0].v = __builtin_amdgcn_wmma_f32_16x16x32_bf16(false, a0.v, false, b0.v, (short)0, acc[0].v, false, false);
        acc[1].v = __builtin_amdgcn_wmma_f32_16x16x32_bf16(false, a0.v, false, b1.v, (short)0, acc[1].v, false, false);
        acc[2].v = __builtin_amdgcn_wmma_f32_16x16x32_bf16(false, a1.v, false, b0.v, (short)0, acc[2].v, false, false);
        acc[3].v = __builtin_amdgcn_wmma_f32_16x16x32_bf16(false, a1.v, false, b1.v, (short)0, acc[3].v, false, false);
    }

    for (int t = 0; t < 4; ++t)
        for (int r = 0; r < 8; ++r)
            atomicAdd(&red[t][(r + hi8) * 16 + l15], acc[t].e[r]);
    __syncthreads();

    for (int i = tid; i < 1024; i += 256) {
        int t = i >> 8, slot = i & 255;
        int row = (t >> 1) * 16 + (slot >> 4);
        int col = (t & 1) * 16 + (slot & 15);
        Am[(size_t)bh * 1024 + row * 32 + col] = red[t][slot] * ATT_SCALE;
    }
}

// ---------------------------------------------------------------------------
// K4: fused  Y = V(512x32)@A(32x32) -> scrambled (64x256) reinterpret (the
// torch reshape) -> F = Y @ Wout + bout -> LayerNorm -> out.
// A staged in LDS pre-swizzled to fragment order (two ds_load_b128/frag);
// Wout B-frags are b128 loads from L2-resident WoutT; LN stats via ds_add_f32.
// ---------------------------------------------------------------------------
__global__ __launch_bounds__(256) void k_out_fused(const __bf16* __restrict__ vb,
                                                   const float* __restrict__ Am,
                                                   const __bf16* __restrict__ WoutT,
                                                   const float* __restrict__ bout,
                                                   const float* __restrict__ gamma,
                                                   const float* __restrict__ beta,
                                                   float* __restrict__ out) {
    __shared__ __bf16 sAf[2 * 32 * 16];   // [ntile][lane][16] fragment order
    __shared__ __bf16 sy[64 * 264];       // Y in scrambled [64][256] layout
    __shared__ float  s_sum[64], s_sq[64];
    const int bh = blockIdx.y, b = bh >> 3, h = bh & 7;
    const int r0 = blockIdx.x * 64;
    const int tid = threadIdx.x, lane = tid & 31, wv = tid >> 5;
    const int l15 = lane & 15, half = lane >> 4, hi8 = half * 8;

    // stage A swizzled into per-lane-contiguous fragment order
    for (int i = 0; i < 4; ++i) {
        int e = i * 256 + tid;            // e = d*32 + c
        int d = e >> 5, c = e & 31;
        int nt = c >> 4;
        int ln = (((d >> 3) & 1) << 4) | (c & 15);
        int j  = (d & 7) + ((d >> 4) << 3);
        sAf[(nt * 32 + ln) * 16 + j] = (__bf16)Am[(size_t)bh * 1024 + e];
    }
    if (tid < 64) { s_sum[tid] = 0.0f; s_sq[tid] = 0.0f; }
    __syncthreads();

    // --- Y = V @ A : 32 Mtiles x 2 Ntiles, single K=32 WMMA each -----------
    const size_t vbase = ((size_t)b * SEQ + (size_t)r0 * 8) * CDIM + h * DH;
    for (int i = 0; i < 8; ++i) {
        int t = wv + 8 * i;               // 0..63
        int mt = t >> 1, nt = t & 1;
        const __bf16* vrow = vb + vbase + (size_t)(mt * 16 + l15) * CDIM;
        AFrag a, bf;
        a.h[0]  = *(const v8bf*)(vrow + hi8);
        a.h[1]  = *(const v8bf*)(vrow + 16 + hi8);
        bf.h[0] = *(const v8bf*)&sAf[(nt * 32 + lane) * 16];
        bf.h[1] = *(const v8bf*)&sAf[(nt * 32 + lane) * 16 + 8];
        CFrag d;
        for (int r = 0; r < 8; ++r) d.e[r] = 0.0f;
        d.v = __builtin_amdgcn_wmma_f32_16x16x32_bf16(
            false, a.v, false, bf.v, (short)0, d.v, false, false);
        for (int r = 0; r < 8; ++r) {
            int vr = mt * 16 + r + hi8;   // 0..511
            int vc = nt * 16 + l15;       // 0..31
            sy[(vr >> 3) * 264 + (vr & 7) * 32 + vc] = (__bf16)d.e[r];
        }
    }
    __syncthreads();

    // --- F = Y(64x256) @ Wout: A-frags ds_load_b128 from sy, B from WoutT --
    const int mt2 = wv >> 1;
    CFrag acc2[8];
    for (int j = 0; j < 8; ++j)
        for (int r = 0; r < 8; ++r) acc2[j].e[r] = 0.0f;
    for (int ks = 0; ks < 8; ++ks) {
        AFrag a;
        const __bf16* yrow = &sy[(mt2 * 16 + l15) * 264 + ks * 32];
        a.h[0] = *(const v8bf*)(yrow + hi8);
        a.h[1] = *(const v8bf*)(yrow + 16 + hi8);
        for (int j = 0; j < 8; ++j) {
            int nt2 = (wv & 1) * 8 + j;
            const __bf16* wrow = WoutT + (size_t)(nt2 * 16 + l15) * CDIM + ks * 32;
            AFrag bf;
            bf.h[0] = *(const v8bf*)(wrow + hi8);
            bf.h[1] = *(const v8bf*)(wrow + 16 + hi8);
            acc2[j].v = __builtin_amdgcn_wmma_f32_16x16x32_bf16(
                false, a.v, false, bf.v, (short)0, acc2[j].v, false, false);
        }
    }

    // --- bias + LayerNorm statistics (ds_add_f32 cross-wave reduce) --------
    float bj[8];
    for (int j = 0; j < 8; ++j)
        bj[j] = bout[((wv & 1) * 8 + j) * 16 + l15];
    for (int r = 0; r < 8; ++r) {
        int row = mt2 * 16 + r + hi8;
        float sm = 0.0f, sq = 0.0f;
        for (int j = 0; j < 8; ++j) {
            float v = acc2[j].e[r] + bj[j];
            sm += v; sq += v * v;
        }
        atomicAdd(&s_sum[row], sm);
        atomicAdd(&s_sq[row],  sq);
    }
    __syncthreads();

    const size_t obase = ((size_t)b * SEQ + (size_t)h * 1024 + r0) * CDIM;
    for (int r = 0; r < 8; ++r) {
        int row = mt2 * 16 + r + hi8;
        float mean = s_sum[row] * (1.0f / 256.0f);
        float var  = s_sq[row] * (1.0f / 256.0f) - mean * mean;
        float rstd = rsqrtf(var + LN_EPS);
        for (int j = 0; j < 8; ++j) {
            int col = ((wv & 1) * 8 + j) * 16 + l15;
            float v = acc2[j].e[r] + bj[j];
            out[obase + (size_t)row * CDIM + col] =
                (v - mean) * rstd * gamma[col] + beta[col];
        }
    }
}

// ---------------------------------------------------------------------------
extern "C" void kernel_launch(void* const* d_in, const int* in_sizes, int n_in,
                              void* d_out, int out_size, void* d_ws, size_t ws_size,
                              hipStream_t stream) {
    (void)in_sizes; (void)n_in; (void)out_size; (void)ws_size;
    const float* x     = (const float*)d_in[0];
    const float* Wqkv  = (const float*)d_in[1];
    const float* Wout  = (const float*)d_in[2];
    const float* bout  = (const float*)d_in[3];
    const float* gamma = (const float*)d_in[4];
    const float* beta  = (const float*)d_in[5];
    float* out = (float*)d_out;
    char* ws = (char*)d_ws;

    __bf16* qT    = (__bf16*)(ws + OFF_Q);
    float*  kT    = (float*)(ws + OFF_K);
    __bf16* vb    = (__bf16*)(ws + OFF_V);
    float*  ms    = (float*)(ws + OFF_M);
    float*  rz    = (float*)(ws + OFF_Z);
    float*  Am    = (float*)(ws + OFF_A);
    __bf16* WqkvT = (__bf16*)(ws + OFF_WQT);
    __bf16* WoutT = (__bf16*)(ws + OFF_WOT);

    k_transpose<<<768, 256, 0, stream>>>(Wqkv, Wout, WqkvT, WoutT);
    k_qkv_gemm <<<dim3(NROWS / 128, 768 / 64), 256, 0, stream>>>(x, WqkvT, qT, kT, vb);
    k_stats    <<<BATCH * CDIM, 256, 0, stream>>>(kT, ms, rz);
    k_amat     <<<BATCH * NHEAD, 256, 0, stream>>>(kT, qT, ms, rz, Am);
    k_out_fused<<<dim3(SEQ / NHEAD / 64, BATCH * NHEAD), 256, 0, stream>>>(
        vb, Am, WoutT, bout, gamma, beta, out);
}
